// AutoAtencao_74887049773586
// MI455X (gfx1250) — matrix operand order repaired and checked by reading, not compiled, not run on hardware
//
#include <hip/hip_runtime.h>

typedef unsigned short u16;
typedef __attribute__((ext_vector_type(16))) __bf16 v16bf;
typedef __attribute__((ext_vector_type(8)))  float  v8f;
typedef __attribute__((ext_vector_type(4)))  u16    u16x4;

#define S_LEN 2048
#define D_DIM 2048
#define H_NUM 16
#define DH 128
#define N3 6144          // 3*D
#define BS 4096          // B*S

union Frag16 {
  v16bf v;
  float4 f4[2];
  u16 u[16];
};

__device__ __forceinline__ u16 f2bf(float f) {
  unsigned int u = __float_as_uint(f);
  u += 0x7FFFu + ((u >> 16) & 1u);      // round to nearest even
  return (u16)(u >> 16);
}

__device__ __forceinline__ v8f wmma_bf16(v16bf a, v16bf b, v8f c) {
  return __builtin_amdgcn_wmma_f32_16x16x32_bf16(false, a, false, b, (short)0, c,
                                                 false, false);
}

// ---- CDNA5 async global->LDS copy (ASYNCcnt-tracked), inline asm ----------
// LDS byte address = low 32 bits of the flat shared-memory address
// (ISA 10.2, LDS aperture).
__device__ __forceinline__ unsigned lds_off(const void* p) {
  return (unsigned)(unsigned long long)p;
}
__device__ __forceinline__ void async_b128(unsigned loff, const u16* g) {
  asm volatile("global_load_async_to_lds_b128 %0, %1, off"
               :: "v"(loff), "v"(g) : "memory");
}
template <int N>
__device__ __forceinline__ void s_wait_async() {
  asm volatile("s_wait_asynccnt %0" :: "i"(N) : "memory");
}

// ---------------------------------------------------------------------------
// fp32 -> bf16 elementwise convert (vectorized x4)
// ---------------------------------------------------------------------------
__global__ void cvt_f32_bf16(const float4* __restrict__ in, u16x4* __restrict__ out,
                             long n4) {
  long i = (long)blockIdx.x * blockDim.x + threadIdx.x;
  if (i < n4) {
    float4 v = in[i];
    u16x4 r;
    r.x = f2bf(v.x); r.y = f2bf(v.y); r.z = f2bf(v.z); r.w = f2bf(v.w);
    out[i] = r;
  }
}

// ---------------------------------------------------------------------------
// fp32 W[K][N] -> bf16 Wt[N][K]  (32x32 LDS tile transpose)
// grid: (N/32, K/32), 256 threads
// ---------------------------------------------------------------------------
__global__ __launch_bounds__(256) void transpose_cvt_w(const float* __restrict__ in,
                                                       u16* __restrict__ out,
                                                       int K, int N) {
  __shared__ u16 t[32][33];
  int n0 = blockIdx.x * 32, k0 = blockIdx.y * 32;
  int r = threadIdx.x >> 5, c = threadIdx.x & 31;
#pragma unroll
  for (int i = 0; i < 4; i++)
    t[r + 8 * i][c] = f2bf(in[(size_t)(k0 + r + 8 * i) * N + n0 + c]);
  __syncthreads();
#pragma unroll
  for (int i = 0; i < 4; i++)
    out[(size_t)(n0 + r + 8 * i) * K + k0 + c] = t[c][r + 8 * i];
}

// ---------------------------------------------------------------------------
// V section of qkv (bf16) -> vt[(b*H+h)*128 + d][S]   (32x32 tile transpose)
// grid: (S/32, B*H*(Dh/32)), 256 threads
// ---------------------------------------------------------------------------
__global__ __launch_bounds__(256) void transpose_v(const u16* __restrict__ qkv,
                                                   u16* __restrict__ vt) {
  __shared__ u16 t[32][33];
  int s0 = blockIdx.x * 32;
  int bh = blockIdx.y >> 2;
  int d0 = (blockIdx.y & 3) * 32;
  int b = bh >> 4, h = bh & 15;
  int r = threadIdx.x >> 5, c = threadIdx.x & 31;
  const u16* src = qkv + (size_t)(b * S_LEN) * N3 + 2 * D_DIM + h * DH;
#pragma unroll
  for (int i = 0; i < 4; i++)
    t[r + 8 * i][c] = src[(size_t)(s0 + r + 8 * i) * N3 + d0 + c];
  __syncthreads();
  u16* dst = vt + (size_t)bh * DH * S_LEN;
#pragma unroll
  for (int i = 0; i < 4; i++)
    dst[(size_t)(d0 + r + 8 * i) * S_LEN + s0 + c] = t[c][r + 8 * i];
}

// ---------------------------------------------------------------------------
// Tiled bf16 WMMA GEMM with async double-buffered LDS staging.
// C[M,N] = A[M,K] * Wt[N,K]^T + bias[N]
// Block tile 128x128, K-step 32; 8 waves, each 32(M) x 64(N) = 2x4 WMMA tiles.
// ---------------------------------------------------------------------------
template <bool OUT_BF16>
__global__ __launch_bounds__(256) void gemm_bf16_kernel(
    const u16* __restrict__ A, const u16* __restrict__ Wt,
    const float* __restrict__ bias, void* __restrict__ Cout,
    int M, int N, int K) {
  __shared__ u16 As[2][128 * 32];   // [row][k]
  __shared__ u16 Ws[2][128 * 32];   // [n][k]

  const int tid  = threadIdx.x;
  const int lane = tid & 31;
  const int w    = tid >> 5;
  const int wm   = w >> 1;
  const int wn   = w & 1;
  const int l16  = lane & 15;
  const int hl   = lane >> 4;
  const int m0   = blockIdx.y * 128;
  const int n0   = blockIdx.x * 128;

  // staging map: 2 threads per row, 32B (16 u16) each
  const int row2 = tid >> 1, ch2 = (tid & 1) * 16;
  const u16* Ag = A  + (size_t)(m0 + row2) * K + ch2;
  const u16* Wg = Wt + (size_t)(n0 + row2) * K + ch2;
  unsigned aoff[2], woff[2];
#pragma unroll
  for (int bfi = 0; bfi < 2; bfi++) {
    aoff[bfi] = lds_off(&As[bfi][row2 * 32 + ch2]);
    woff[bfi] = lds_off(&Ws[bfi][row2 * 32 + ch2]);
  }

  v8f acc[2][4];
#pragma unroll
  for (int i = 0; i < 2; i++)
#pragma unroll
    for (int j = 0; j < 4; j++) acc[i][j] = (v8f)0.0f;

  auto stage = [&](int bfi, int k0) {
    async_b128(aoff[bfi],      Ag + k0);
    async_b128(aoff[bfi] + 16, Ag + k0 + 8);
    async_b128(woff[bfi],      Wg + k0);
    async_b128(woff[bfi] + 16, Wg + k0 + 8);
  };

  stage(0, 0);
  int buf = 0;
  for (int k0 = 0; k0 < K; k0 += 32, buf ^= 1) {
    if (k0 + 32 < K) {
      stage(buf ^ 1, k0 + 32);   // prefetch next tile while current computes
      s_wait_async<4>();         // current tile's 4 copies have landed
    } else {
      s_wait_async<0>();
    }
    __syncthreads();

    Frag16 af[2], bfr[4];
#pragma unroll
    for (int mt = 0; mt < 2; mt++) {
      const u16* p = &As[buf][(wm * 32 + mt * 16 + l16) * 32 + hl * 16];
      af[mt].f4[0] = *(const float4*)p;          // K = hl*16 + 0..7
      af[mt].f4[1] = *(const float4*)(p + 8);    // K = hl*16 + 8..15
    }
#pragma unroll
    for (int nt = 0; nt < 4; nt++) {
      const u16* p = &Ws[buf][(wn * 64 + nt * 16 + l16) * 32 + hl * 16];
      bfr[nt].f4[0] = ((const float4*)p)[0];     // K = hl*16 + 0..7
      bfr[nt].f4[1] = ((const float4*)p)[1];     // K = hl*16 + 8..15
    }
#pragma unroll
    for (int mt = 0; mt < 2; mt++)
#pragma unroll
      for (int nt = 0; nt < 4; nt++)
        acc[mt][nt] = wmma_bf16(af[mt].v, bfr[nt].v, acc[mt][nt]);
    __syncthreads();
  }

  // epilogue: bias + store (C layout: M = r + 8*hl, N = l16)
#pragma unroll
  for (int mt = 0; mt < 2; mt++) {
    int rowbase = m0 + wm * 32 + mt * 16 + hl * 8;
#pragma unroll
    for (int nt = 0; nt < 4; nt++) {
      int col = n0 + wn * 64 + nt * 16 + l16;
      float bv = bias[col];
#pragma unroll
      for (int r = 0; r < 8; r++) {
        float v = acc[mt][nt][r] + bv;
        if (OUT_BF16)
          ((u16*)Cout)[(size_t)(rowbase + r) * N + col] = f2bf(v);
        else
          ((float*)Cout)[(size_t)(rowbase + r) * N + col] = v;
      }
    }
  }
}

// ---------------------------------------------------------------------------
// Flash attention (causal), async double-buffered K/V staging.
// qkv bf16 [BS][6144]; vt bf16 [(b*H+h)*128 + d][S]; out bf16 [BS][2048].
// Grid (S/128, B*H), 256 threads (8 waves); each wave owns 16 query rows.
// ---------------------------------------------------------------------------
__global__ __launch_bounds__(256) void attn_kernel(const u16* __restrict__ qkv,
                                                   const u16* __restrict__ vt,
                                                   u16* __restrict__ out) {
  __shared__ u16 Ks[2][32 * 128];   // [key][d]
  __shared__ u16 Vs[2][128 * 32];   // [d][key]
  __shared__ u16 Ps[8 * 16 * 32];   // per-wave P scratch [M][keycol]

  const int tid  = threadIdx.x;
  const int lane = tid & 31;
  const int w    = tid >> 5;
  const int l16  = lane & 15;
  const int hl   = lane >> 4;
  const int b    = blockIdx.y >> 4;
  const int h    = blockIdx.y & 15;
  const int q0   = blockIdx.x * 128;
  const int qw0  = q0 + w * 16;

  const u16* base = qkv + (size_t)(b * S_LEN) * N3;

  // staging maps
  const int kkey = tid >> 3, kseg = (tid & 7) * 16;          // K tile
  const u16* Kg = base + (size_t)kkey * N3 + D_DIM + h * DH + kseg;
  const int vd = tid >> 1, vh = (tid & 1) * 16;              // V tile
  const u16* Vg = vt + ((size_t)blockIdx.y * DH + vd) * S_LEN + vh;
  unsigned koff[2], voff[2];
#pragma unroll
  for (int bfi = 0; bfi < 2; bfi++) {
    koff[bfi] = lds_off(&Ks[bfi][kkey * 128 + kseg]);
    voff[bfi] = lds_off(&Vs[bfi][vd * 32 + vh]);
  }
  auto stage = [&](int bfi, int kb0) {
    const u16* kp = Kg + (size_t)kb0 * N3;
    async_b128(koff[bfi],      kp);
    async_b128(koff[bfi] + 16, kp + 8);
    const u16* vp = Vg + kb0;
    async_b128(voff[bfi],      vp);
    async_b128(voff[bfi] + 16, vp + 8);
  };

  // preload Q fragments (A layout), 4 k-steps of 32 over Dh=128
  Frag16 qf[4];
  {
    const u16* qp = base + (size_t)(qw0 + l16) * N3 + h * DH;
#pragma unroll
    for (int kk = 0; kk < 4; kk++) {
      int cb = kk * 32 + hl * 16;
      qf[kk].f4[0] = *(const float4*)(qp + cb);
      qf[kk].f4[1] = *(const float4*)(qp + cb + 8);
    }
  }

  v8f o[8];
#pragma unroll
  for (int i = 0; i < 8; i++) o[i] = (v8f)0.0f;
  float mrow[8], lrow[8];
#pragma unroll
  for (int r = 0; r < 8; r++) { mrow[r] = -1e30f; lrow[r] = 0.0f; }

  const float sc = 0.08838834764831845f;  // 1/sqrt(128)
  const int kbmax = q0 + 128;

  stage(0, 0);
  int buf = 0;
  for (int kb0 = 0; kb0 < kbmax; kb0 += 32, buf ^= 1) {
    if (kb0 + 32 < kbmax) {
      stage(buf ^ 1, kb0 + 32);
      s_wait_async<4>();
    } else {
      s_wait_async<0>();
    }
    __syncthreads();

    if (kb0 <= qw0 + 15) {
      // scores: 16 x 32 = two 16x16 tiles, K over Dh
      v8f s0 = (v8f)0.0f, s1 = (v8f)0.0f;
#pragma unroll
      for (int kk = 0; kk < 4; kk++) {
        Frag16 b0, b1;
        int kboff = kk * 32 + hl * 16;
        const u16* p0 = &Ks[buf][l16 * 128 + kboff];
        const u16* p1 = &Ks[buf][(16 + l16) * 128 + kboff];
        b0.f4[0] = ((const float4*)p0)[0]; b0.f4[1] = ((const float4*)p0)[1];
        b1.f4[0] = ((const float4*)p1)[0]; b1.f4[1] = ((const float4*)p1)[1];
        s0 = wmma_bf16(qf[kk].v, b0.v, s0);
        s1 = wmma_bf16(qf[kk].v, b1.v, s1);
      }
      // causal mask + scale, per-row max
      int key0 = kb0 + l16, key1 = key0 + 16;
      float rmax[8];
#pragma unroll
      for (int r = 0; r < 8; r++) {
        int q = qw0 + r + hl * 8;
        float v0 = (key0 <= q) ? s0[r] * sc : -1e30f;
        float v1 = (key1 <= q) ? s1[r] * sc : -1e30f;
        s0[r] = v0; s1[r] = v1;
        rmax[r] = fmaxf(v0, v1);
      }
#pragma unroll
      for (int msk = 1; msk <= 8; msk <<= 1)
#pragma unroll
        for (int r = 0; r < 8; r++)
          rmax[r] = fmaxf(rmax[r], __shfl_xor(rmax[r], msk, 32));

      // online softmax update
      float alpha[8], rsum[8];
#pragma unroll
      for (int r = 0; r < 8; r++) {
        float mn = fmaxf(mrow[r], rmax[r]);
        alpha[r] = __expf(mrow[r] - mn);
        mrow[r] = mn;
        float p0 = __expf(s0[r] - mn);
        float p1 = __expf(s1[r] - mn);
        s0[r] = p0; s1[r] = p1;
        rsum[r] = p0 + p1;
      }
#pragma unroll
      for (int msk = 1; msk <= 8; msk <<= 1)
#pragma unroll
        for (int r = 0; r < 8; r++)
          rsum[r] += __shfl_xor(rsum[r], msk, 32);
#pragma unroll
      for (int r = 0; r < 8; r++) lrow[r] = lrow[r] * alpha[r] + rsum[r];
#pragma unroll
      for (int nt = 0; nt < 8; nt++)
#pragma unroll
        for (int r = 0; r < 8; r++) o[nt][r] *= alpha[r];

      // repack P (C layout) -> A layout via per-wave LDS scratch
      u16* pw = Ps + w * 16 * 32;
#pragma unroll
      for (int r = 0; r < 8; r++) {
        int Mi = r + hl * 8;
        pw[Mi * 32 + l16]      = f2bf(s0[r]);
        pw[Mi * 32 + 16 + l16] = f2bf(s1[r]);
      }
      Frag16 pf;
      {
        const u16* pp = pw + l16 * 32 + hl * 16;
        pf.f4[0] = *(const float4*)pp;
        pf.f4[1] = *(const float4*)(pp + 8);
      }
      // PV: 8 output 16x16 tiles across Dh
#pragma unroll
      for (int nt = 0; nt < 8; nt++) {
        Frag16 vb;
        const u16* vp = &Vs[buf][(nt * 16 + l16) * 32 + hl * 16];
        vb.f4[0] = ((const float4*)vp)[0];
        vb.f4[1] = ((const float4*)vp)[1];
        o[nt] = wmma_bf16(pf.v, vb.v, o[nt]);
      }
    }
    __syncthreads();
  }

  // normalize + store bf16 [token][h*128 + d]
  u16* ob = out + (size_t)(b * S_LEN + qw0) * D_DIM + h * DH;
#pragma unroll
  for (int r = 0; r < 8; r++) {
    float inv = 1.0f / lrow[r];
    int Mi = r + hl * 8;
#pragma unroll
    for (int nt = 0; nt < 8; nt++)
      ob[(size_t)Mi * D_DIM + nt * 16 + l16] = f2bf(o[nt][r] * inv);
  }
}

// ---------------------------------------------------------------------------
extern "C" void kernel_launch(void* const* d_in, const int* in_sizes, int n_in,
                              void* d_out, int out_size, void* d_ws, size_t ws_size,
                              hipStream_t stream) {
  const float* x     = (const float*)d_in[0];
  const float* W_in  = (const float*)d_in[1];
  const float* b_in  = (const float*)d_in[2];
  const float* W_out = (const float*)d_in[3];
  const float* b_out = (const float*)d_in[4];
  float* outp        = (float*)d_out;

  u16* x_bf   = (u16*)d_ws;                                   // BS*D
  u16* winT   = x_bf  + (size_t)BS * D_DIM;                   // N3*D (=[N][K])
  u16* woutT  = winT  + (size_t)N3 * D_DIM;                   // D*D  (=[N][K])
  u16* qkv    = woutT + (size_t)D_DIM * D_DIM;                // BS*N3
  u16* vt     = qkv   + (size_t)BS * N3;                      // B*H*DH*S
  u16* attn_o = vt    + (size_t)2 * H_NUM * DH * S_LEN;       // BS*D

  {
    long n4 = (long)BS * D_DIM / 4;
    cvt_f32_bf16<<<(int)((n4 + 255) / 256), 256, 0, stream>>>(
        (const float4*)x, (u16x4*)x_bf, n4);
  }
  transpose_cvt_w<<<dim3(N3 / 32, D_DIM / 32), 256, 0, stream>>>(W_in, winT,
                                                                 D_DIM, N3);
  transpose_cvt_w<<<dim3(D_DIM / 32, D_DIM / 32), 256, 0, stream>>>(W_out, woutT,
                                                                    D_DIM, D_DIM);

  // QKV projection: [4096,2048] x [2048,6144] -> bf16 qkv (+ b_in)
  gemm_bf16_kernel<true><<<dim3(N3 / 128, BS / 128), 256, 0, stream>>>(
      x_bf, winT, b_in, (void*)qkv, BS, N3, D_DIM);

  // V transpose per (b,h): qkv V section -> vt
  transpose_v<<<dim3(S_LEN / 32, 2 * H_NUM * (DH / 32)), 256, 0, stream>>>(qkv, vt);

  // Flash attention
  attn_kernel<<<dim3(S_LEN / 128, 2 * H_NUM), 256, 0, stream>>>(qkv, vt, attn_o);

  // Output projection: [4096,2048] x [2048,2048] -> fp32 d_out (+ b_out)
  gemm_bf16_kernel<false><<<dim3(D_DIM / 128, BS / 128), 256, 0, stream>>>(
      attn_o, woutT, b_out, (void*)outp, BS, D_DIM, D_DIM);
}
